// Attention_19035295056537
// MI455X (gfx1250) — matrix operand order repaired and checked by reading, not compiled, not run on hardware
//
#include <hip/hip_runtime.h>
#include <hip/hip_bf16.h>
#include <stdint.h>
#include <limits.h>

typedef __attribute__((ext_vector_type(16))) _Float16 v16h;
typedef __attribute__((ext_vector_type(8)))  _Float16 v8h;
typedef __attribute__((ext_vector_type(8)))  float    v8f;
typedef __attribute__((ext_vector_type(4)))  unsigned int u32x4;
typedef __attribute__((ext_vector_type(8)))  int          i32x8;
typedef __attribute__((ext_vector_type(4)))  int          i32x4;

#define EMBED  1024
#define HEADS  16
#define HDIM   64
#define BATCH  2
#define SEQ    2048
#define NTOK   (BATCH * SEQ)
#define QKVW   (3 * EMBED)

#define WMMA_F16(a, b, c) \
    __builtin_amdgcn_wmma_f32_16x16x32_f16(false, (a), false, (b), (short)0, (c), false, false)

#ifndef __has_builtin
#define __has_builtin(x) 0
#endif
#if __has_builtin(__builtin_amdgcn_tensor_load_to_lds) && \
    __has_builtin(__builtin_amdgcn_s_wait_tensorcnt)
#define HAVE_TDM 1
#else
#define HAVE_TDM 0
#endif

// ---------------------------------------------------------------------------
// Fragment loaders matching CDNA5 16-bit WMMA VGPR layouts (CDNA5 ISA)
// A (16x32, MxK): M = lane&15, half h -> K = (h%8) + 16*(h/8) + 8*(lane>=16)
// B (32x16, KxN) stored col-major [n][k]: N = lane&15, half h -> K = (h%8) + 16*(h/8) + 8*(lane>=16)
// C/D (16x16): vgpr r -> M = r + 8*(lane>=16), N = lane&15
// ---------------------------------------------------------------------------
__device__ __forceinline__ v16h load_frag_a(const _Float16* p, int rowbase, int ld, int kbase) {
    int lane = threadIdx.x & 31;
    int row  = rowbase + (lane & 15);
    int koff = (lane & 16) ? 8 : 0;
    const _Float16* q = p + (size_t)row * ld + kbase + koff;
    v8h lo = *(const v8h*)(q);        // K = koff+0..7
    v8h hi = *(const v8h*)(q + 16);   // K = 16+koff+0..7
    v16h r;
#pragma unroll
    for (int i = 0; i < 8; i++) { r[i] = lo[i]; r[i + 8] = hi[i]; }
    return r;
}

__device__ __forceinline__ v16h load_frag_b(const _Float16* p, int nbase, int ld, int kbase) {
    int lane = threadIdx.x & 31;
    int n    = nbase + (lane & 15);
    int koff = (lane & 16) ? 8 : 0;
    const _Float16* q = p + (size_t)n * ld + kbase + koff;
    v8h lo = *(const v8h*)(q);        // K = koff+0..7
    v8h hi = *(const v8h*)(q + 16);   // K = 16+koff+0..7
    v16h r;
#pragma unroll
    for (int i = 0; i < 8; i++) { r[i] = lo[i]; r[i + 8] = hi[i]; }
    return r;
}

// ---------------------------------------------------------------------------
// Tensor Data Mover: 2D tile (rows x cols halfs, row stride in halfs) -> LDS
// D# bitfields per CDNA5 ISA
// ---------------------------------------------------------------------------
#if HAVE_TDM
__device__ __forceinline__ void tdm_load_2d(uint32_t lds_off, const _Float16* gptr,
                                            int rows, int cols, int row_stride) {
    uint64_t ga = (uint64_t)(uintptr_t)gptr;
    u32x4 g0;
    g0[0] = 1u;                                    // count=1, is_restore=0, gather off
    g0[1] = lds_off;                               // lds_addr (bytes)
    g0[2] = (uint32_t)ga;                          // global_addr[31:0]
    g0[3] = ((uint32_t)(ga >> 32) & 0x01FFFFFFu)   // global_addr[56:32]
            | 0x80000000u;                         // type=2 ("image")
    i32x8 g1;
    g1[0] = 0x00010000;                            // wg_mask=0, data_size=1 (2 bytes)
    g1[1] = (cols & 0xFFFF) << 16;                 // tensor_dim0[15:0]
    g1[2] = (rows & 0xFFFF) << 16;                 // tensor_dim0[31:16]=0 | tensor_dim1[15:0]
    g1[3] = (cols & 0xFFFF) << 16;                 // tensor_dim1[31:16]=0 | tile_dim0
    g1[4] = (rows & 0xFFFF);                       // tile_dim1 | tile_dim2=0
    g1[5] = row_stride;                            // tensor_dim0_stride[31:0]
    g1[6] = 0;
    g1[7] = 0;
    i32x4 z4 = {0, 0, 0, 0};
#if __clang_major__ >= 23
    i32x8 z8 = {0, 0, 0, 0, 0, 0, 0, 0};
    __builtin_amdgcn_tensor_load_to_lds(g0, g1, z4, z4, z8, 0);
#else
    __builtin_amdgcn_tensor_load_to_lds(g0, g1, z4, z4, 0);
#endif
}
#endif

// ---------------------------------------------------------------------------
// f32 -> f16 conversion
// ---------------------------------------------------------------------------
__global__ __launch_bounds__(256) void cvt_kernel(const float* __restrict__ in,
                                                  _Float16* __restrict__ out, int n) {
    for (int i = blockIdx.x * 256 + threadIdx.x; i < n; i += gridDim.x * 256)
        out[i] = (_Float16)in[i];
}

// ---------------------------------------------------------------------------
// 256x128 tile GEMM, f16 in, f32 accumulate, double-buffered software pipeline.
// 256 threads = 8 waves; each wave owns two 16-row strips -> 16 WMMAs/k-step.
// A tile (256x32) prefetched by the Tensor Data Mover; B tile prefetched to
// registers before compute and drained into LDS (transposed) after.
// ---------------------------------------------------------------------------
template <bool F16OUT>
__global__ __launch_bounds__(256) void gemm_f16_256x128(
    const _Float16* __restrict__ A, const _Float16* __restrict__ B,
    void* __restrict__ Cout, int M, int N, int K)
{
    __shared__ __align__(16) _Float16 sA[2][256 * 32];   // 2 x 16 KB, [m][k]
    __shared__ __align__(16) _Float16 sBt[2][128 * 32];  // 2 x 8 KB,  [n][k]
    int tid  = threadIdx.x;
    int lane = tid & 31, wave = tid >> 5;
    int mb = blockIdx.y * 256, nb = blockIdx.x * 128;
    int nsteps = K >> 5;

    v8f acc[2][8] = {};

    // ---- prologue: stage k-step 0 into buffer 0
#if HAVE_TDM
    if (wave == 0)
        tdm_load_2d((uint32_t)(uintptr_t)sA[0], A + (size_t)mb * K, 256, 32, K);
#else
#pragma unroll
    for (int t = 0; t < 4; t++) {
        int e = (tid + t * 256) * 8;
        int m = e >> 5, k = e & 31;
        *(v8h*)(sA[0] + e) = *(const v8h*)(A + (size_t)(mb + m) * K + k);
    }
#endif
#pragma unroll
    for (int t = 0; t < 2; t++) {
        int e = (tid + t * 256) * 8;
        int k = e >> 7, n = e & 127;
        v8h d = *(const v8h*)(B + (size_t)k * N + nb + n);
#pragma unroll
        for (int i = 0; i < 8; i++) sBt[0][(n + i) * 32 + k] = d[i];
    }
#if HAVE_TDM
    if (wave == 0) __builtin_amdgcn_s_wait_tensorcnt(0);
#endif
    __syncthreads();

    for (int s = 0; s < nsteps; s++) {
        int cur = s & 1, nxt = cur ^ 1;
        int k0n = (s + 1) << 5;
        bool more = (s + 1 < nsteps);

        // ---- prefetch next tile: TDM for A, global->reg for B
        v8h breg[2];
        if (more) {
#if HAVE_TDM
            if (wave == 0)
                tdm_load_2d((uint32_t)(uintptr_t)sA[nxt],
                            A + (size_t)mb * K + k0n, 256, 32, K);
#else
#pragma unroll
            for (int t = 0; t < 4; t++) {
                int e = (tid + t * 256) * 8;
                int m = e >> 5, k = e & 31;
                *(v8h*)(sA[nxt] + e) =
                    *(const v8h*)(A + (size_t)(mb + m) * K + k0n + k);
            }
#endif
#pragma unroll
            for (int t = 0; t < 2; t++) {
                int e = (tid + t * 256) * 8;
                int k = e >> 7, n = e & 127;
                breg[t] = *(const v8h*)(B + (size_t)(k0n + k) * N + nb + n);
            }
        }

        // ---- compute on current buffers (overlaps TDM + global B loads)
        v16h a0 = load_frag_a(sA[cur], wave * 32, 32, 0);
        v16h a1 = load_frag_a(sA[cur], wave * 32 + 16, 32, 0);
#pragma unroll
        for (int nt = 0; nt < 8; nt++) {
            v16h b = load_frag_b(sBt[cur], nt * 16, 32, 0);
            acc[0][nt] = WMMA_F16(a0, b, acc[0][nt]);
            acc[1][nt] = WMMA_F16(a1, b, acc[1][nt]);
        }

        // ---- drain prefetched B into the next LDS buffer
        if (more) {
#pragma unroll
            for (int t = 0; t < 2; t++) {
                int e = (tid + t * 256) * 8;
                int k = e >> 7, n = e & 127;
#pragma unroll
                for (int i = 0; i < 8; i++) sBt[nxt][(n + i) * 32 + k] = breg[t][i];
            }
#if HAVE_TDM
            if (wave == 0) __builtin_amdgcn_s_wait_tensorcnt(0);
#endif
        }
        __syncthreads();
    }

    int cbase = nb + (lane & 15);
#pragma unroll
    for (int mt = 0; mt < 2; mt++) {
        int rbase = mb + wave * 32 + mt * 16 + ((lane & 16) ? 8 : 0);
#pragma unroll
        for (int nt = 0; nt < 8; nt++) {
#pragma unroll
            for (int r = 0; r < 8; r++) {
                size_t off = (size_t)(rbase + r) * N + cbase + nt * 16;
                if (F16OUT) ((_Float16*)Cout)[off] = (_Float16)acc[mt][nt][r];
                else        ((float*)Cout)[off]    = acc[mt][nt][r];
            }
        }
    }
}

// ---------------------------------------------------------------------------
// RoPE applied in place to q,k sections of the f16 qkv buffer.
// ---------------------------------------------------------------------------
__global__ __launch_bounds__(256) void rope_kernel(_Float16* __restrict__ qkv) {
    int idx = blockIdx.x * 256 + threadIdx.x;
    int i   = idx & 31;
    int h   = (idx >> 5) & 15;
    int tok = idx >> 9;
    if (tok >= NTOK) return;
    int s = tok & (SEQ - 1);
    float inv = __powf(10000.f, -(float)i * (1.0f / 32.0f));
    float sn, cs;
    __sincosf((float)s * inv, &sn, &cs);
    size_t base = (size_t)tok * QKVW + h * HDIM + i;
    float q1 = (float)qkv[base], q2 = (float)qkv[base + 32];
    qkv[base]        = (_Float16)(q1 * cs - q2 * sn);
    qkv[base + 32]   = (_Float16)(q2 * cs + q1 * sn);
    float k1 = (float)qkv[base + EMBED], k2 = (float)qkv[base + EMBED + 32];
    qkv[base + EMBED]      = (_Float16)(k1 * cs - k2 * sn);
    qkv[base + EMBED + 32] = (_Float16)(k2 * cs + k1 * sn);
}

// ---------------------------------------------------------------------------
// Flash-style attention, double-buffered. grid = (SEQ/64, BATCH*HEADS),
// 128 threads = 4 waves. K tile j+1 is loaded via TDM after block j computes;
// V tile staged with b128 reads + transposing b16 stores. One barrier/block.
// Masking is branchless: per-lane integer limit, mask iff col > limit.
// ---------------------------------------------------------------------------
__global__ __launch_bounds__(128) void attn_kernel(
    const _Float16* __restrict__ qkv, _Float16* __restrict__ y,
    const int* __restrict__ is_causal_p, const unsigned char* __restrict__ amask)
{
    __shared__ __align__(16) _Float16 sK[2][32 * 64];   // [kv][d] (col-major B for Q*K^T)
    __shared__ __align__(16) _Float16 sVt[2][64 * 32];  // [d][kv] (col-major B for P*V)
    __shared__ __align__(16) _Float16 sP[4 * 16 * 32];  // per-wave P tile [row][kv]

    int lane = threadIdx.x & 31, wave = threadIdx.x >> 5;
    int bh = blockIdx.y;
    int b = bh >> 4, h = bh & 15;
    int qbase  = blockIdx.x * 64;
    int causal = is_causal_p[0];
    int rowmul = causal ? 1 : 0;      // uniform: limit = rowmul*row + limbase

    const _Float16* qptr = qkv + (size_t)b * SEQ * QKVW + h * HDIM;
    v16h qa0 = load_frag_a(qptr, qbase + wave * 16, QKVW, 0);
    v16h qa1 = load_frag_a(qptr, qbase + wave * 16, QKVW, 32);

    v8f o[4] = {};
    float mrow[8], lrow[8];
#pragma unroll
    for (int r = 0; r < 8; r++) { mrow[r] = -1e30f; lrow[r] = 0.f; }

    int jend = causal ? (qbase + 64) : SEQ;
    int nj = jend >> 5;

    // ---- prologue: stage kv-block 0 into buffer 0
    {
        size_t kvrow0 = (size_t)(b * SEQ) * QKVW + h * HDIM;
#if HAVE_TDM
        if (wave == 0)
            tdm_load_2d((uint32_t)(uintptr_t)sK[0], qkv + kvrow0 + EMBED, 32, 64, QKVW);
#else
        for (int t = threadIdx.x; t < 2048; t += 128) {
            int kv = t >> 6, d = t & 63;
            sK[0][t] = qkv[kvrow0 + (size_t)kv * QKVW + EMBED + d];
        }
#endif
#pragma unroll
        for (int t = 0; t < 2; t++) {
            int e = (threadIdx.x + t * 128) * 8;    // 0..2040 step 8
            int kv = e >> 6, d = e & 63;
            v8h dv = *(const v8h*)(qkv + kvrow0 + (size_t)kv * QKVW + 2 * EMBED + d);
#pragma unroll
            for (int i = 0; i < 8; i++) sVt[0][(d + i) * 32 + kv] = dv[i];
        }
#if HAVE_TDM
        if (wave == 0) __builtin_amdgcn_s_wait_tensorcnt(0);
#endif
        __syncthreads();
    }

    for (int s = 0; s < nj; s++) {
        int jb = s << 5;
        int cur = s & 1, nxt = cur ^ 1;
        bool more = (s + 1 < nj);

        // ---- scores: 2 tiles (kv 0-15, 16-31), K-dim 64 over head dim
        v8f s0 = {}, s1 = {};
        {
            v16h kb;
            kb = load_frag_b(sK[cur], 0, 64, 0);   s0 = WMMA_F16(qa0, kb, s0);
            kb = load_frag_b(sK[cur], 0, 64, 32);  s0 = WMMA_F16(qa1, kb, s0);
            kb = load_frag_b(sK[cur], 16, 64, 0);  s1 = WMMA_F16(qa0, kb, s1);
            kb = load_frag_b(sK[cur], 16, 64, 32); s1 = WMMA_F16(qa1, kb, s1);
        }

        _Float16* pp = sP + wave * 512;
        int rbase = qbase + wave * 16 + ((lane & 16) ? 8 : 0);
        int c0 = jb + (lane & 15), c1 = c0 + 16;

        // Per-lane mask limits, computed once per kv-block (no in-loop branches):
        //   causal:     limit = row          (mask iff col > row)
        //   non-causal: limit = INT_MIN/MAX  (always / never masked)
        int lim0 = 0, lim1 = 0;
        if (!causal) {
            lim0 = (amask[b * SEQ + c0] == 0) ? INT_MIN : INT_MAX;
            lim1 = (amask[b * SEQ + c1] == 0) ? INT_MIN : INT_MAX;
        }

#pragma unroll
        for (int r = 0; r < 8; r++) {
            int row  = rbase + r;
            int rsel = rowmul * row;                  // causal: row, else 0
            float v0 = (c0 > lim0 + rsel) ? -1e30f : s0[r] * 0.125f;
            float v1 = (c1 > lim1 + rsel) ? -1e30f : s1[r] * 0.125f;
            float tm = fmaxf(v0, v1);
            tm = fmaxf(tm, __shfl_xor(tm, 1));
            tm = fmaxf(tm, __shfl_xor(tm, 2));
            tm = fmaxf(tm, __shfl_xor(tm, 4));
            tm = fmaxf(tm, __shfl_xor(tm, 8));
            float mn   = fmaxf(mrow[r], tm);
            float corr = __expf(mrow[r] - mn);
            float p0 = __expf(v0 - mn), p1 = __expf(v1 - mn);
            float rs = p0 + p1;
            rs += __shfl_xor(rs, 1);
            rs += __shfl_xor(rs, 2);
            rs += __shfl_xor(rs, 4);
            rs += __shfl_xor(rs, 8);
            lrow[r] = lrow[r] * corr + rs;
            mrow[r] = mn;
#pragma unroll
            for (int t2 = 0; t2 < 4; t2++) o[t2][r] *= corr;
            int prl = r + ((lane & 16) ? 8 : 0);
            pp[prl * 32 + (lane & 15)]      = (_Float16)p0;
            pp[prl * 32 + 16 + (lane & 15)] = (_Float16)p1;
        }
        // P store->load is intra-wave (DS is in-order per wave); stop compiler reorder
        asm volatile("" ::: "memory");

        // ---- O += P (16x32) x V (32x64)
        v16h pa = load_frag_a(pp, 0, 32, 0);
#pragma unroll
        for (int t2 = 0; t2 < 4; t2++) {
            v16h vb = load_frag_b(sVt[cur], t2 * 16, 32, 0);
            o[t2] = WMMA_F16(pa, vb, o[t2]);
        }

        // ---- prefetch kv-block j+1 into the other buffers
        if (more) {
            size_t kvrowN = (size_t)(b * SEQ + jb + 32) * QKVW + h * HDIM;
#if HAVE_TDM
            if (wave == 0)
                tdm_load_2d((uint32_t)(uintptr_t)sK[nxt], qkv + kvrowN + EMBED,
                            32, 64, QKVW);
#else
            for (int t = threadIdx.x; t < 2048; t += 128) {
                int kv = t >> 6, d = t & 63;
                sK[nxt][t] = qkv[kvrowN + (size_t)kv * QKVW + EMBED + d];
            }
#endif
#pragma unroll
            for (int t = 0; t < 2; t++) {
                int e = (threadIdx.x + t * 128) * 8;
                int kv = e >> 6, d = e & 63;
                v8h dv = *(const v8h*)(qkv + kvrowN + (size_t)kv * QKVW + 2 * EMBED + d);
#pragma unroll
                for (int i = 0; i < 8; i++) sVt[nxt][(d + i) * 32 + kv] = dv[i];
            }
#if HAVE_TDM
            if (wave == 0) __builtin_amdgcn_s_wait_tensorcnt(0);
#endif
        }
        __syncthreads();
    }

    int rbase = qbase + wave * 16 + ((lane & 16) ? 8 : 0);
#pragma unroll
    for (int t2 = 0; t2 < 4; t2++) {
#pragma unroll
        for (int r = 0; r < 8; r++) {
            float val = o[t2][r] / lrow[r];
            y[(size_t)(b * SEQ + rbase + r) * EMBED + h * HDIM + t2 * 16 + (lane & 15)] =
                (_Float16)val;
        }
    }
}

// ---------------------------------------------------------------------------
// Host-side orchestration
// ---------------------------------------------------------------------------
extern "C" void kernel_launch(void* const* d_in, const int* in_sizes, int n_in,
                              void* d_out, int out_size, void* d_ws, size_t ws_size,
                              hipStream_t stream) {
    const float*          x     = (const float*)d_in[0];
    const unsigned char*  amask = (const unsigned char*)d_in[1];
    const float*          wqkv  = (const float*)d_in[2];
    const float*          wout  = (const float*)d_in[3];
    const int*            isc   = (const int*)d_in[4];

    char* ws = (char*)d_ws;
    const size_t MB = 1024 * 1024;
    _Float16* xh    = (_Float16*)(ws);                 // 8 MB  (4096x1024)
    _Float16* wqkvh = (_Float16*)(ws + 8 * MB);        // 6 MB  (1024x3072)
    _Float16* wouth = (_Float16*)(ws + 14 * MB);       // 2 MB  (1024x1024)
    _Float16* qkvh  = (_Float16*)(ws + 16 * MB);       // 24 MB (4096x3072)
    _Float16* yh    = (_Float16*)(ws + 40 * MB);       // 8 MB  (4096x1024)

    cvt_kernel<<<2048, 256, 0, stream>>>(x, xh, NTOK * EMBED);
    cvt_kernel<<<2048, 256, 0, stream>>>(wqkv, wqkvh, EMBED * QKVW);
    cvt_kernel<<<1024, 256, 0, stream>>>(wout, wouth, EMBED * EMBED);

    // qkv = x @ w_qkv  (f16 out)
    gemm_f16_256x128<true><<<dim3(QKVW / 128, NTOK / 256), 256, 0, stream>>>(
        xh, wqkvh, (void*)qkvh, NTOK, QKVW, EMBED);

    // RoPE on q,k in place
    rope_kernel<<<(NTOK * HEADS * 32) / 256, 256, 0, stream>>>(qkvh);

    // attention -> yh (f16)
    attn_kernel<<<dim3(SEQ / 64, BATCH * HEADS), 128, 0, stream>>>(qkvh, yh, isc, amask);

    // out = y @ w_out  (f32 out)
    gemm_f16_256x128<false><<<dim3(EMBED / 128, NTOK / 256), 256, 0, stream>>>(
        yh, wouth, (void*)d_out, NTOK, EMBED, EMBED);
}
